// RealNet_18940805775535
// MI455X (gfx1250) — matrix operand order repaired and checked by reading, not compiled, not run on hardware
//
#include <hip/hip_runtime.h>
#include <hip/hip_bf16.h>

// ---------------------------------------------------------------------------
// RealNet recurrence on gfx1250 (MI455X):
//   h <- RMSNorm(tanh(h @ W + B)) * norm_weight, 16 sequential steps.
// GEMM uses V_WMMA_F32_16X16X32_BF16 with bf16x3 compensation (W_hi/W_lo,
// h_hi/h_lo splits) for ~fp32 accuracy. W is pre-packed into the WMMA
// B-fragment layout; each (K-chunk x 64-col) B tile is a contiguous 4KB slab
// staged into LDS with GLOBAL_LOAD_ASYNC_TO_LDS_B128 (double-buffered,
// ASYNCcnt-tracked). K-loop is kept un-unrolled so the WMMA accumulators
// stay register-pinned (in-place D==C accumulation, no v_mov copies).
// ---------------------------------------------------------------------------

#define NN     4096
#define BATCH  256
#define NIO    16
#define STEPS  16
#define EPSF   1.1920929e-07f

typedef __bf16 bf16_t;
typedef bf16_t bf16x16 __attribute__((ext_vector_type(16)));
typedef float  floatx8 __attribute__((ext_vector_type(8)));

__device__ __forceinline__ unsigned short f32_to_bf16_rne(float f) {
    union { float f; unsigned int u; } c; c.f = f;
    unsigned int u = c.u;
    unsigned int r = u + 0x7FFFu + ((u >> 16) & 1u);   // round-to-nearest-even
    return (unsigned short)(r >> 16);
}
__device__ __forceinline__ float bf16_bits_to_f32(unsigned short b) {
    union { unsigned int u; float f; } c; c.u = ((unsigned int)b) << 16;
    return c.f;
}

// One per-lane 16B async copy, global -> LDS, tracked on ASYNCcnt.
__device__ __forceinline__ void async_b128_to_lds(const void* lds_ptr,
                                                  unsigned goff_bytes,
                                                  const unsigned int* gbase) {
    unsigned lds = (unsigned)(uintptr_t)lds_ptr;   // low 32 bits = LDS offset
    asm volatile("global_load_async_to_lds_b128 %0, %1, %2"
                 :
                 : "v"(lds), "v"(goff_bytes), "s"(gbase)
                 : "memory");
}

// ---------------------------------------------------------------------------
// Pack W (f32, row-major [K=4096][N=4096]) into WMMA B-fragment order as
// bf16 hi/lo splits. Dword index = (kchunk*NN + n)*16 + p, p = half*8 + j,
// holding the pair (W[k][n], W[k+1][n]) with k = kchunk*32 + half*16 + 2j.
// ---------------------------------------------------------------------------
__global__ void convert_w_kernel(const float* __restrict__ W,
                                 unsigned int* __restrict__ Wp_hi,
                                 unsigned int* __restrict__ Wp_lo) {
    unsigned int idx = blockIdx.x * blockDim.x + threadIdx.x;  // NN*NN/2 dwords
    unsigned int p  = idx & 15u;
    unsigned int n  = (idx >> 4) & (NN - 1);
    unsigned int kc = idx >> 16;
    unsigned int half = p >> 3, j = p & 7u;
    unsigned int k = kc * 32u + half * 16u + 2u * j;

    float w0 = W[(size_t)k * NN + n];
    float w1 = W[(size_t)(k + 1) * NN + n];
    unsigned short h0 = f32_to_bf16_rne(w0);
    unsigned short h1 = f32_to_bf16_rne(w1);
    unsigned short l0 = f32_to_bf16_rne(w0 - bf16_bits_to_f32(h0));
    unsigned short l1 = f32_to_bf16_rne(w1 - bf16_bits_to_f32(h1));
    Wp_hi[idx] = ((unsigned int)h1 << 16) | h0;
    Wp_lo[idx] = ((unsigned int)l1 << 16) | l0;
}

// ---------------------------------------------------------------------------
// Step 0: h0 == 0, so signal = B + inj (x with input_pos cols scaled).
// ---------------------------------------------------------------------------
__global__ void step0_kernel(const float* __restrict__ x,
                             const float* __restrict__ Bv,
                             const float* __restrict__ isc,
                             const float* __restrict__ osc,
                             const float* __restrict__ nw,
                             const int*   __restrict__ ipos,
                             const int*   __restrict__ opos,
                             unsigned short* __restrict__ h_hi,
                             unsigned short* __restrict__ h_lo,
                             float* __restrict__ out) {
    const int row = blockIdx.x;
    const int tid = threadIdx.x;
    float a[16];
    float ss = 0.f;
#pragma unroll
    for (int i = 0; i < 16; ++i) {
        int col = tid + i * 256;
        float inj = x[(size_t)row * NN + col];
        for (int q = 0; q < NIO; ++q) if (col == ipos[q]) inj *= isc[q];
        float v = tanhf(Bv[col] + inj);
        a[i] = v;
        ss += v * v;
    }
    __shared__ float red[256];
    red[tid] = ss;
    __syncthreads();
    for (int s = 128; s > 0; s >>= 1) {
        if (tid < s) red[tid] += red[tid + s];
        __syncthreads();
    }
    float inv = rsqrtf(red[0] / (float)NN + EPSF);
#pragma unroll
    for (int i = 0; i < 16; ++i) {
        int col = tid + i * 256;
        float h = a[i] * inv * nw[col];
        unsigned short hb = f32_to_bf16_rne(h);
        h_hi[(size_t)row * NN + col] = hb;
        h_lo[(size_t)row * NN + col] = f32_to_bf16_rne(h - bf16_bits_to_f32(hb));
        float o = h;
        for (int q = 0; q < NIO; ++q) if (col == opos[q]) o *= osc[q];
        out[((size_t)row * STEPS + 0) * NN + col] = o;
    }
}

// ---------------------------------------------------------------------------
// GEMM + tanh with async-LDS-staged B operand (double-buffered).
// Block = 256 threads = 8 waves; 64x64 output tile; wave: 16 rows x 32 cols.
// ---------------------------------------------------------------------------
__global__ __launch_bounds__(256) void gemm_tanh_kernel(
        const unsigned short* __restrict__ h_hi,
        const unsigned short* __restrict__ h_lo,
        const unsigned int*  __restrict__ Wp_hi,
        const unsigned int*  __restrict__ Wp_lo,
        const float* __restrict__ Bv,
        float* __restrict__ a_buf,
        float* __restrict__ partial) {
    const int tid   = threadIdx.x;
    const int lane  = tid & 31;
    const int wid   = tid >> 5;
    const int half  = lane >> 4;
    const int l16   = lane & 15;
    const int mtile = wid & 3;
    const int npair = wid >> 2;
    const int rowbase = blockIdx.y * 64 + mtile * 16;
    const unsigned nblock = blockIdx.x * 64;

    __shared__ __align__(16) unsigned int smem[2][2][1024];  // 16KB B stage

    floatx8 acc0 = {0.f, 0.f, 0.f, 0.f, 0.f, 0.f, 0.f, 0.f};
    floatx8 acc1 = {0.f, 0.f, 0.f, 0.f, 0.f, 0.f, 0.f, 0.f};

    const int arow = rowbase + l16;
    const unsigned short* pa_hi = h_hi + (size_t)arow * NN;
    const unsigned short* pa_lo = h_lo + (size_t)arow * NN;

    const unsigned goff0 = (nblock * 16u + (unsigned)tid * 4u) * 4u;
    const unsigned fidx0 = (unsigned)(npair * 32 + l16) * 16u + half * 8u;
    const unsigned fidx1 = (unsigned)(npair * 32 + 16 + l16) * 16u + half * 8u;

    async_b128_to_lds(&smem[0][0][tid * 4], goff0, Wp_hi);
    async_b128_to_lds(&smem[0][1][tid * 4], goff0, Wp_lo);

#pragma unroll 1
    for (int kc = 0; kc < NN / 32; ++kc) {
        if (kc + 1 < NN / 32) {
            const unsigned goff = goff0 + (unsigned)(kc + 1) * (NN * 64u);
            unsigned int (*nb)[1024] = smem[(kc + 1) & 1];
            async_b128_to_lds(&nb[0][tid * 4], goff, Wp_hi);
            async_b128_to_lds(&nb[1][tid * 4], goff, Wp_lo);
            asm volatile("s_wait_asynccnt 0x2" ::: "memory");
        } else {
            asm volatile("s_wait_asynccnt 0x0" ::: "memory");
        }
        __syncthreads();   // chunk kc visible block-wide

        union Frag { bf16x16 v; uint4 q[2]; } ah, al, bh0, bl0, bh1, bl1;
        const int ka = kc * 32 + half * 16;
        ah.q[0] = *(const uint4*)(pa_hi + ka);
        ah.q[1] = *(const uint4*)(pa_hi + ka + 8);
        al.q[0] = *(const uint4*)(pa_lo + ka);
        al.q[1] = *(const uint4*)(pa_lo + ka + 8);

        const unsigned int* bufh = smem[kc & 1][0];
        const unsigned int* bufl = smem[kc & 1][1];
        bh0.q[0] = *(const uint4*)(bufh + fidx0);
        bh0.q[1] = *(const uint4*)(bufh + fidx0 + 4);
        bl0.q[0] = *(const uint4*)(bufl + fidx0);
        bl0.q[1] = *(const uint4*)(bufl + fidx0 + 4);
        bh1.q[0] = *(const uint4*)(bufh + fidx1);
        bh1.q[1] = *(const uint4*)(bufh + fidx1 + 4);
        bl1.q[0] = *(const uint4*)(bufl + fidx1);
        bl1.q[1] = *(const uint4*)(bufl + fidx1 + 4);

        acc0 = __builtin_amdgcn_wmma_f32_16x16x32_bf16(false, ah.v, false, bh0.v,
                                                       (short)0, acc0, false, false);
        acc0 = __builtin_amdgcn_wmma_f32_16x16x32_bf16(false, al.v, false, bh0.v,
                                                       (short)0, acc0, false, false);
        acc0 = __builtin_amdgcn_wmma_f32_16x16x32_bf16(false, ah.v, false, bl0.v,
                                                       (short)0, acc0, false, false);
        acc1 = __builtin_amdgcn_wmma_f32_16x16x32_bf16(false, ah.v, false, bh1.v,
                                                       (short)0, acc1, false, false);
        acc1 = __builtin_amdgcn_wmma_f32_16x16x32_bf16(false, al.v, false, bh1.v,
                                                       (short)0, acc1, false, false);
        acc1 = __builtin_amdgcn_wmma_f32_16x16x32_bf16(false, ah.v, false, bl1.v,
                                                       (short)0, acc1, false, false);

        __syncthreads();   // reads of buf[kc&1] done before overwrite
    }

    // Epilogue: + bias, tanh, store activations, per-row sum of squares.
    // C/D layout: element j -> row = rowbase + half*8 + j.
    float ssj[8];
#pragma unroll
    for (int j = 0; j < 8; ++j) ssj[j] = 0.f;

    {
        const int col = nblock + npair * 32 + l16;
        const float bias = Bv[col];
#pragma unroll
        for (int j = 0; j < 8; ++j) {
            const int r = rowbase + half * 8 + j;
            float v = tanhf(acc0[j] + bias);
            a_buf[(size_t)r * NN + col] = v;
            ssj[j] += v * v;
        }
    }
    {
        const int col = nblock + npair * 32 + 16 + l16;
        const float bias = Bv[col];
#pragma unroll
        for (int j = 0; j < 8; ++j) {
            const int r = rowbase + half * 8 + j;
            float v = tanhf(acc1[j] + bias);
            a_buf[(size_t)r * NN + col] = v;
            ssj[j] += v * v;
        }
    }

    __shared__ float sm[64][2];
#pragma unroll
    for (int j = 0; j < 8; ++j) {
        float s = ssj[j];
        s += __shfl_xor(s, 1, 32);
        s += __shfl_xor(s, 2, 32);
        s += __shfl_xor(s, 4, 32);
        s += __shfl_xor(s, 8, 32);
        if (l16 == 0) sm[mtile * 16 + half * 8 + j][npair] = s;
    }
    __syncthreads();
    if (tid < 64) {
        const int r = blockIdx.y * 64 + tid;
        partial[(size_t)r * (NN / 64) + blockIdx.x] = sm[tid][0] + sm[tid][1];
    }
}

// ---------------------------------------------------------------------------
// Normalize (deterministic LDS-tree over the 64 per-blockN partials).
// ---------------------------------------------------------------------------
__global__ void norm_kernel(const float* __restrict__ a_buf,
                            const float* __restrict__ partial,
                            const float* __restrict__ nw,
                            const float* __restrict__ osc,
                            const int*   __restrict__ opos,
                            unsigned short* __restrict__ h_hi,
                            unsigned short* __restrict__ h_lo,
                            float* __restrict__ out,
                            int t,
                            float* __restrict__ hfinal) {
    const int idx = blockIdx.x * 256 + threadIdx.x;
    const int row = idx >> 12;            // NN == 4096
    const int col = idx & (NN - 1);
    const int tid = threadIdx.x;

    __shared__ float pr[64];
    if (tid < 64) pr[tid] = partial[(size_t)row * 64 + tid];
    __syncthreads();
    for (int s = 32; s > 0; s >>= 1) {
        if (tid < s) pr[tid] += pr[tid + s];
        __syncthreads();
    }
    const float inv = rsqrtf(pr[0] / (float)NN + EPSF);

    float h = a_buf[idx] * inv * nw[col];
    unsigned short hb = f32_to_bf16_rne(h);
    h_hi[idx] = hb;
    h_lo[idx] = f32_to_bf16_rne(h - bf16_bits_to_f32(hb));

    float o = h;
    for (int q = 0; q < NIO; ++q) if (col == opos[q]) o *= osc[q];
    out[((size_t)row * STEPS + t) * NN + col] = o;
    if (hfinal) hfinal[idx] = h;
}

// ---------------------------------------------------------------------------
// Launcher: pack W once, run step 0, then 15 x (gemm -> norm), all on stream.
// Workspace: Wp_hi(32MB) | Wp_lo(32MB) | h_hi(2MB) | h_lo(2MB) | a_buf(4MB)
//            | partial(64KB)   ~= 72.1 MB
// ---------------------------------------------------------------------------
extern "C" void kernel_launch(void* const* d_in, const int* in_sizes, int n_in,
                              void* d_out, int out_size, void* d_ws, size_t ws_size,
                              hipStream_t stream) {
    const float* x    = (const float*)d_in[0];
    const float* W    = (const float*)d_in[1];
    const float* Bv   = (const float*)d_in[2];
    const float* isc  = (const float*)d_in[3];
    const float* osc  = (const float*)d_in[4];
    const float* nw   = (const float*)d_in[5];
    const int*   ipos = (const int*)d_in[6];
    const int*   opos = (const int*)d_in[7];

    float* out    = (float*)d_out;
    float* hfinal = out + (size_t)BATCH * STEPS * NN;

    unsigned int*  Wp_hi   = (unsigned int*)d_ws;
    unsigned int*  Wp_lo   = Wp_hi + (size_t)NN * NN / 2;
    unsigned short* h_hi   = (unsigned short*)(Wp_lo + (size_t)NN * NN / 2);
    unsigned short* h_lo   = h_hi + (size_t)BATCH * NN;
    float*          a_buf  = (float*)(h_lo + (size_t)BATCH * NN);
    float*          prtl   = a_buf + (size_t)BATCH * NN;

    convert_w_kernel<<<(NN * NN / 2) / 256, 256, 0, stream>>>(W, Wp_hi, Wp_lo);

    step0_kernel<<<BATCH, 256, 0, stream>>>(x, Bv, isc, osc, nw, ipos, opos,
                                            h_hi, h_lo, out);

    dim3 ggrid(NN / 64, BATCH / 64);
    for (int t = 1; t < STEPS; ++t) {
        gemm_tanh_kernel<<<ggrid, 256, 0, stream>>>(h_hi, h_lo, Wp_hi, Wp_lo,
                                                    Bv, a_buf, prtl);
        norm_kernel<<<(BATCH * NN) / 256, 256, 0, stream>>>(
            a_buf, prtl, nw, osc, opos, h_hi, h_lo, out, t,
            (t == STEPS - 1) ? hfinal : nullptr);
    }
}
